// MultiHeadAttentionHalf_22333829939583
// MI455X (gfx1250) — matrix operand order repaired and checked by reading, not compiled, not run on hardware
//
#include <hip/hip_runtime.h>
#include <hip/hip_bf16.h>

// ---------------------------------------------------------------------------
// MHA forward for B=4,S=1024,E=1024,H=16,D=64 using gfx1250 WMMA (bf16 -> f32).
// Outputs: d_out[0..4095]            = context_vector [B,E]
//          d_out[4096..4096+4M-1]    = attn_weights   [B,S,S] (mean over heads)
// ---------------------------------------------------------------------------

#define Bb 4
#define Ss 1024
#define Ee 1024
#define Hh 16
#define Dd 64

typedef __attribute__((ext_vector_type(16))) __bf16 v16bf;
typedef __attribute__((ext_vector_type(8)))  float  v8f;

union FragU {
    v16bf v;
    uint4 u[2];
};

static __device__ __forceinline__ v8f wmma_bf16(v16bf a, v16bf b, v8f c) {
    return __builtin_amdgcn_wmma_f32_16x16x32_bf16(
        /*neg_a=*/false, a, /*neg_b=*/false, b,
        /*c_mod=*/(short)0, c, /*reuse_a=*/false, /*reuse_b=*/false);
}

// convert 8 consecutive f32 at p into dst[o..o+7] as bf16
static __device__ __forceinline__ void cvt8(const float* __restrict__ p, v16bf& v, int o) {
    float4 q0 = reinterpret_cast<const float4*>(p)[0];
    float4 q1 = reinterpret_cast<const float4*>(p)[1];
    v[o+0] = (__bf16)q0.x; v[o+1] = (__bf16)q0.y;
    v[o+2] = (__bf16)q0.z; v[o+3] = (__bf16)q0.w;
    v[o+4] = (__bf16)q1.x; v[o+5] = (__bf16)q1.y;
    v[o+6] = (__bf16)q1.z; v[o+7] = (__bf16)q1.w;
}

// ---------------------------------------------------------------------------
// Kernel 0: bulk f32 -> bf16 conversion (8 elements per thread, b128 I/O).
// ---------------------------------------------------------------------------
__global__ __launch_bounds__(256) void cvt_bf16_kernel(
    const float* __restrict__ src, __bf16* __restrict__ dst)
{
    const int i = blockIdx.x * 256 + threadIdx.x;     // one uint4 of bf16 out
    const float4* s = reinterpret_cast<const float4*>(src) + (size_t)i * 2;
    float4 q0 = s[0], q1 = s[1];
    union { __bf16 h[8]; uint4 u; } t;
    t.h[0] = (__bf16)q0.x; t.h[1] = (__bf16)q0.y;
    t.h[2] = (__bf16)q0.z; t.h[3] = (__bf16)q0.w;
    t.h[4] = (__bf16)q1.x; t.h[5] = (__bf16)q1.y;
    t.h[6] = (__bf16)q1.z; t.h[7] = (__bf16)q1.w;
    reinterpret_cast<uint4*>(dst)[i] = t.u;
}

// ---------------------------------------------------------------------------
// Kernel 1: QKV = X @ Wqkv^T + bias.  M=4096, N=3072, K=1024, bf16 inputs.
// Writes Q (scaled by 1/8) and K as bf16 [B,H,S,D], V transposed bf16 [B,H,D,S].
// Block = 128 threads (4 waves); each wave computes a 32x64 strip
// (2 A-frags share 4 B-frags -> 8 WMMAs per K-step, pure load+wmma loop).
// ---------------------------------------------------------------------------
__global__ __launch_bounds__(128) void qkv_kernel(
    const __bf16* __restrict__ X, const __bf16* __restrict__ W,
    const float* __restrict__ bias,
    __bf16* __restrict__ Qw, __bf16* __restrict__ Kw, __bf16* __restrict__ Vt)
{
    const int lane = threadIdx.x & 31;
    const int wave = threadIdx.x >> 5;
    const int half = lane >> 4;
    const int lm   = lane & 15;
    const int m0   = blockIdx.y * 128 + wave * 32;  // this wave's 32 rows
    const int n0   = blockIdx.x * 64;               // 4 column tiles of 16

    v8f c[2][4] = {{v8f{}, v8f{}, v8f{}, v8f{}},
                   {v8f{}, v8f{}, v8f{}, v8f{}}};

    for (int kk = 0; kk < Ee; kk += 32) {
        // A fragments: rows m0+lm and m0+16+lm; K = kk+half*8+{0..7}, kk+16+half*8+{0..7}
        FragU a0, a1;
        const __bf16* ap0 = X + (m0 + lm) * Ee + kk + half * 8;
        a0.u[0] = *reinterpret_cast<const uint4*>(ap0);
        a0.u[1] = *reinterpret_cast<const uint4*>(ap0 + 16);
        const __bf16* ap1 = ap0 + 16 * Ee;
        a1.u[0] = *reinterpret_cast<const uint4*>(ap1);
        a1.u[1] = *reinterpret_cast<const uint4*>(ap1 + 16);
        #pragma unroll
        for (int j = 0; j < 4; ++j) {
            // B fragment: col n, K = kk+half*8+{0..7}, kk+16+half*8+{0..7}
            const __bf16* bp = W + (n0 + j * 16 + lm) * Ee + kk + half * 8;
            FragU bf;
            bf.u[0] = *reinterpret_cast<const uint4*>(bp);
            bf.u[1] = *reinterpret_cast<const uint4*>(bp + 16);
            c[0][j] = wmma_bf16(a0.v, bf.v, c[0][j]);
            c[1][j] = wmma_bf16(a1.v, bf.v, c[1][j]);
        }
    }

    // Scatter with bias; C layout: lane col n = lm, VGPR r -> row r + 8*half.
    #pragma unroll
    for (int mi = 0; mi < 2; ++mi) {
        #pragma unroll
        for (int j = 0; j < 4; ++j) {
            const int n  = n0 + j * 16 + lm;
            const float bv = bias[n];
            #pragma unroll
            for (int r = 0; r < 8; ++r) {
                const int m  = m0 + mi * 16 + r + half * 8;
                const int bb = m >> 10, s = m & 1023;
                float val = c[mi][j][r] + bv;
                if (n < Ee) {                       // Q, fold in 1/sqrt(64)
                    const int h = n >> 6, d = n & 63;
                    Qw[(((bb * Hh + h) * Ss + s) * Dd) + d] = (__bf16)(val * 0.125f);
                } else if (n < 2 * Ee) {            // K
                    const int n2 = n - Ee, h = n2 >> 6, d = n2 & 63;
                    Kw[(((bb * Hh + h) * Ss + s) * Dd) + d] = (__bf16)val;
                } else {                            // V transposed [B,H,D,S]
                    const int n2 = n - 2 * Ee, h = n2 >> 6, d = n2 & 63;
                    Vt[(((bb * Hh + h) * Dd + d) * Ss) + s] = (__bf16)val;
                }
            }
        }
    }
}

// ---------------------------------------------------------------------------
// Kernel 2: fused attention. One wave per block; block = (b, 16-row slice).
// Loops over 16 heads: scores (WMMA) -> softmax (LDS) -> ctx (WMMA).
// Accumulates head-mean attn in LDS (deterministic), ctx column sums to ws.
// Dynamic LDS: 2 * 16 * 1024 floats = 128 KB.
// ---------------------------------------------------------------------------
__global__ __launch_bounds__(32) void attn_kernel(
    const __bf16* __restrict__ Qw, const __bf16* __restrict__ Kw,
    const __bf16* __restrict__ Vt,
    float* __restrict__ ctxpart,      // [B, 64, 1024]
    float* __restrict__ out)          // d_out
{
    extern __shared__ float lds[];
    float* attn  = lds;               // [16][1024] current head probs
    float* meanw = lds + 16 * Ss;     // [16][1024] mean over heads

    const int lane = threadIdx.x;
    const int half = lane >> 4;
    const int lm   = lane & 15;
    const int b    = blockIdx.x >> 6;
    const int sblk = blockIdx.x & 63;
    const int srow = sblk * 16 + lm;

    for (int i = lane; i < 16 * Ss; i += 32) meanw[i] = 0.0f;
    __syncthreads();

    for (int h = 0; h < Hh; ++h) {
        const __bf16* Qh = Qw + (size_t)((b * Hh + h) * Ss) * Dd;
        const __bf16* Kh = Kw + (size_t)((b * Hh + h) * Ss) * Dd;
        const __bf16* Vh = Vt + (size_t)((b * Hh + h) * Dd) * Ss;

        // Q A-fragments for this wave's 16 rows (d-chunks 0..31 and 32..63)
        FragU a0, a1;
        {
            const __bf16* q = Qh + srow * Dd;
            a0.u[0] = *reinterpret_cast<const uint4*>(q +  0 + half * 8);
            a0.u[1] = *reinterpret_cast<const uint4*>(q + 16 + half * 8);
            a1.u[0] = *reinterpret_cast<const uint4*>(q + 32 + half * 8);
            a1.u[1] = *reinterpret_cast<const uint4*>(q + 48 + half * 8);
        }

        // ---- scores: 16 rows x 1024 keys ----
        for (int t0 = 0; t0 < Ss; t0 += 16) {
            const __bf16* kp = Kh + (t0 + lm) * Dd + half * 8;
            FragU b0, b1;
            b0.u[0] = *reinterpret_cast<const uint4*>(kp);
            b0.u[1] = *reinterpret_cast<const uint4*>(kp + 16);
            b1.u[0] = *reinterpret_cast<const uint4*>(kp + 32);
            b1.u[1] = *reinterpret_cast<const uint4*>(kp + 48);
            v8f cs = {};
            cs = wmma_bf16(a0.v, b0.v, cs);
            cs = wmma_bf16(a1.v, b1.v, cs);
            #pragma unroll
            for (int r = 0; r < 8; ++r)
                attn[(r + half * 8) * Ss + t0 + lm] = cs[r];
        }
        __syncthreads();

        // ---- softmax: lane pair (lm, lm+16) handles row lm, 512 cols each ----
        {
            float* ar = attn + lm * Ss + half * 512;
            float mx = -3.402823466e38f;
            for (int i = 0; i < 512; ++i) mx = fmaxf(mx, ar[i]);
            mx = fmaxf(mx, __shfl_xor(mx, 16, 32));
            float sum = 0.0f;
            for (int i = 0; i < 512; ++i) { float e = __expf(ar[i] - mx); ar[i] = e; sum += e; }
            sum += __shfl_xor(sum, 16, 32);
            const float inv = 1.0f / sum;
            float* mr = meanw + lm * Ss + half * 512;
            for (int i = 0; i < 512; ++i) {
                float p = ar[i] * inv;
                ar[i] = p;
                mr[i] += p * 0.0625f;     // 1/16 heads
            }
        }
        __syncthreads();

        // ---- ctx = attn[16,1024] @ V[1024,64]  (B from Vt rows, contiguous) ----
        v8f cc[4] = {v8f{}, v8f{}, v8f{}, v8f{}};
        for (int tk = 0; tk < Ss; tk += 32) {
            FragU a;
            const float* arow = attn + lm * Ss + tk + half * 8;
            cvt8(arow,      a.v, 0);
            cvt8(arow + 16, a.v, 8);
            #pragma unroll
            for (int j = 0; j < 4; ++j) {
                const __bf16* vp = Vh + (j * 16 + lm) * Ss + tk + half * 8;
                FragU bf;
                bf.u[0] = *reinterpret_cast<const uint4*>(vp);
                bf.u[1] = *reinterpret_cast<const uint4*>(vp + 16);
                cc[j] = wmma_bf16(a.v, bf.v, cc[j]);
            }
        }
        // column sums over this block's 16 rows -> ctxpart[b][sblk][h*64+d]
        #pragma unroll
        for (int j = 0; j < 4; ++j) {
            float ps = 0.0f;
            #pragma unroll
            for (int r = 0; r < 8; ++r) ps += cc[j][r];
            ps += __shfl_xor(ps, 16, 32);
            if (half == 0)
                ctxpart[(b * 64 + sblk) * Ee + h * Dd + j * 16 + lm] = ps;
        }
        __syncthreads();
    }

    // write head-mean attention weights: d_out[4096 + b*S*S + s*S + t]
    float* dst = out + Bb * Ee + (size_t)(b * Ss + sblk * 16) * Ss;
    for (int i = lane; i < 16 * Ss; i += 32) dst[i] = meanw[i];
}

// ---------------------------------------------------------------------------
// Kernel 3a: reduce ctx partial sums over the 64 row-blocks -> mean over S.
// ---------------------------------------------------------------------------
__global__ __launch_bounds__(256) void ctx_reduce_kernel(
    const float* __restrict__ part, float* __restrict__ ctxmean)
{
    const int idx = blockIdx.x * 256 + threadIdx.x;   // 4096 = B*E
    const int b = idx >> 10, k = idx & 1023;
    float s = 0.0f;
    for (int blk = 0; blk < 64; ++blk)
        s += part[(b * 64 + blk) * Ee + k];
    ctxmean[idx] = s * (1.0f / (float)Ss);
}

// ---------------------------------------------------------------------------
// Kernel 3b: context_vector[b,e] = ctxmean[b,:] . Wout[e,:] + bias[e]
// (mean over S commutes through the linear layer)
// ---------------------------------------------------------------------------
__global__ __launch_bounds__(256) void outproj_kernel(
    const float* __restrict__ ctxmean, const float* __restrict__ Wout,
    const float* __restrict__ bias, float* __restrict__ out)
{
    const int idx = blockIdx.x * 256 + threadIdx.x;   // 4096 = B*E
    const int b = idx >> 10, e = idx & 1023;
    const float* x = ctxmean + b * Ee;
    const float* w = Wout + e * Ee;
    float s = bias[e];
    for (int k = 0; k < Ee; k += 4) {
        s += x[k]   * w[k]   + x[k+1] * w[k+1]
           + x[k+2] * w[k+2] + x[k+3] * w[k+3];
    }
    out[idx] = s;
}

// ---------------------------------------------------------------------------
extern "C" void kernel_launch(void* const* d_in, const int* in_sizes, int n_in,
                              void* d_out, int out_size, void* d_ws, size_t ws_size,
                              hipStream_t stream)
{
    const float* lstm_out  = (const float*)d_in[0];   // [B,S,E]
    const float* w_in      = (const float*)d_in[1];   // [3E,E]
    const float* b_in      = (const float*)d_in[2];   // [3E]
    const float* w_out     = (const float*)d_in[3];   // [E,E]
    const float* b_out     = (const float*)d_in[4];   // [E]
    float* out = (float*)d_out;

    // workspace layout
    const size_t nX   = (size_t)Bb * Ss * Ee;         // 4,194,304
    const size_t nW   = (size_t)3 * Ee * Ee;          // 3,145,728
    const size_t nQKV = (size_t)Bb * Hh * Ss * Dd;    // 4,194,304 per tensor
    __bf16* Xb = (__bf16*)d_ws;
    __bf16* Wb = Xb + nX;
    __bf16* Qw = Wb + nW;
    __bf16* Kw = Qw + nQKV;
    __bf16* Vt = Kw + nQKV;
    float*  ctxpart = (float*)(Vt + nQKV);            // [B,64,E] = 1 MB
    float*  ctxmean = ctxpart + (size_t)Bb * 64 * Ee; // [B,E]

    // 0) one-time f32 -> bf16 conversion of GEMM inputs
    cvt_bf16_kernel<<<(int)(nX / 8 / 256), 256, 0, stream>>>(lstm_out, Xb);
    cvt_bf16_kernel<<<(int)(nW / 8 / 256), 256, 0, stream>>>(w_in, Wb);

    // 1) QKV projection (bf16 WMMA GEMM, scale folded into Q)
    qkv_kernel<<<dim3(3 * Ee / 64, (Bb * Ss) / 128), 128, 0, stream>>>(
        Xb, Wb, b_in, Qw, Kw, Vt);

    // 2) fused attention: scores + softmax + head-mean + ctx column sums
    attn_kernel<<<Bb * (Ss / 16), 32, 2 * 16 * Ss * sizeof(float), stream>>>(
        Qw, Kw, Vt, ctxpart, out);

    // 3) mean over S, then tiny GEMV for context_vector
    ctx_reduce_kernel<<<(Bb * Ee) / 256, 256, 0, stream>>>(ctxpart, ctxmean);
    outproj_kernel<<<(Bb * Ee) / 256, 256, 0, stream>>>(ctxmean, w_out, b_out, out);
}
